// Decoder_33088428049154
// MI455X (gfx1250) — compile-verified, matrix operand order checked by reading, not run on hardware
//
#include <hip/hip_runtime.h>
#include <hip/hip_bf16.h>
#include <math.h>

// ---------------------------------------------------------------------------
// Shapes (fixed by the reference)
// ---------------------------------------------------------------------------
#define BB   4
#define TT   128
#define UU   64
#define DD   512
#define HH   4
#define DH   128
#define DFF  2048
#define JJ   512
#define ODIM 4096
#define LL   6

typedef __attribute__((ext_vector_type(16))) _Float16 v16h;
typedef __attribute__((ext_vector_type(8)))  _Float16 v8h;
typedef __attribute__((ext_vector_type(8)))  float    v8f;

// Constant-modifier wrapper: WMMA builtin modifiers must be ICEs.
__device__ __forceinline__ v8f wmma_f16(const v16h& a, const v16h& b, const v8f& c)
{
    return __builtin_amdgcn_wmma_f32_16x16x32_f16(
        false, a, false, b, (short)0, c, false, false);
}

// ---------------------------------------------------------------------------
// Generic WMMA GEMM:  C[M,N] = epilogue( A[M,K] @ B[K,N] )
//   flags bit0: ReLU
//   flags bit1: += residual (ldc layout)
//   flags bit2: joint mode — A[r,k] = tanh(z_enc[r/U, k] + z_dec[(b*U+u), k])
//
// 128x128 tile per 128-thread block (4 waves), K stepped by 32.
// Each wave owns two 16-row strips; per K-step: 2 A-frag + 8 B-frag loads
// (ds_load_b128 pairs) feed 16 v_wmma_f32_16x16x32_f16 (each B fragment is
// reused by two accumulators), doubling the WMMA-per-LDS-load ratio vs a
// 64x64 tile. Tiles are staged in LDS pre-swizzled into the per-lane WMMA
// fragment layout (ISA 16-bit A/B layouts): one contiguous 32B v16h read per
// fragment.
// ---------------------------------------------------------------------------
#define TILE_M 128
#define TILE_N 128
#define TILE_K 32

__global__ void __launch_bounds__(128)
gemm_f16wmma(const float* __restrict__ A,
             const float* __restrict__ Bm,
             const float* __restrict__ bias,
             const float* __restrict__ residual,
             float* __restrict__ C,
             const float* __restrict__ z_enc,
             const float* __restrict__ z_dec,
             int M, int N, int K,
             int lda, int ldb, int ldc,
             int flags)
{
    // [row-strip / n-subtile][lane][16 halfs] — each lane slot 32B, 32B aligned
    __shared__ __align__(32) _Float16 Afrag[8][32][16];
    __shared__ __align__(32) _Float16 Bfrag[8][32][16];

    const int tid     = threadIdx.x;
    const int lane    = tid & 31;
    const int wave    = tid >> 5;
    const int rowBase = blockIdx.y * TILE_M;
    const int colBase = blockIdx.x * TILE_N;

    v8f acc[16] = {};   // [nt*2 + strip]

    for (int k0 = 0; k0 < K; k0 += TILE_K) {
        // ---- stage A: 128 rows x 4 groups of 8 cols; one b128 LDS store each
#pragma unroll
        for (int it = 0; it < 4; ++it) {
            const int w  = tid + it * 128;      // 0..511
            const int r  = w >> 2;              // row in tile (0..127)
            const int hi = w & 3;               // 8-col group
            const int gk = k0 + hi * 8;
            float f[8];
            if (flags & 4) {
                const int grow = rowBase + r;   // ((b*T + t)*U + u)
                const int bt   = grow >> 6;     // / U
                const int u    = grow & 63;     // % U
                const int b    = bt >> 7;       // / T
                const float4* pe = (const float4*)&z_enc[(size_t)bt * JJ + gk];
                const float4* pd = (const float4*)&z_dec[(size_t)(b * UU + u) * JJ + gk];
                const float4 e0 = pe[0], e1 = pe[1];
                const float4 d0 = pd[0], d1 = pd[1];
                f[0] = tanhf(e0.x + d0.x); f[1] = tanhf(e0.y + d0.y);
                f[2] = tanhf(e0.z + d0.z); f[3] = tanhf(e0.w + d0.w);
                f[4] = tanhf(e1.x + d1.x); f[5] = tanhf(e1.y + d1.y);
                f[6] = tanhf(e1.z + d1.z); f[7] = tanhf(e1.w + d1.w);
            } else {
                const float4* p = (const float4*)&A[(size_t)(rowBase + r) * lda + gk];
                const float4 a0 = p[0], a1 = p[1];
                f[0] = a0.x; f[1] = a0.y; f[2] = a0.z; f[3] = a0.w;
                f[4] = a1.x; f[5] = a1.y; f[6] = a1.z; f[7] = a1.w;
            }
            v8h hv;
#pragma unroll
            for (int i = 0; i < 8; ++i) hv[i] = (_Float16)f[i];
            // ISA A layout: hi=0 -> lanes 0-15 idx0; hi=1 -> lanes 16-31 idx0;
            //               hi=2 -> lanes 0-15 idx8; hi=3 -> lanes 16-31 idx8
            const int strip  = r >> 4;
            const int lane_d = (r & 15) + ((hi & 1) << 4);
            const int idx    = (hi >> 1) << 3;
            *(v8h*)&Afrag[strip][lane_d][idx] = hv;
        }

        // ---- stage B: 128 cols x 4 groups of 8 k; coalesced across lanes
#pragma unroll
        for (int it = 0; it < 4; ++it) {
            const int w   = tid + it * 128;     // 0..511
            const int col = w & 127;
            const int rg  = w >> 7;             // 8-k group
            const int gc  = colBase + col;
            const float* p = &Bm[(size_t)(k0 + rg * 8) * ldb + gc];
            v8h hv;
#pragma unroll
            for (int i = 0; i < 8; ++i)
                hv[i] = (_Float16)p[(size_t)i * ldb];
            const int nt     = col >> 4;
            const int lane_d = (col & 15) + ((rg & 1) << 4);
            const int idx    = (rg >> 1) << 3;
            *(v8h*)&Bfrag[nt][lane_d][idx] = hv;
        }
        __syncthreads();

        // ---- compute: 2 A fragments per wave, 8 B fragments, 16 WMMAs
        const v16h af0 = *(const v16h*)&Afrag[wave][lane][0];
        const v16h af1 = *(const v16h*)&Afrag[wave + 4][lane][0];
#pragma unroll
        for (int nt = 0; nt < 8; ++nt) {
            const v16h bf = *(const v16h*)&Bfrag[nt][lane][0];
            acc[nt * 2 + 0] = wmma_f16(af0, bf, acc[nt * 2 + 0]);
            acc[nt * 2 + 1] = wmma_f16(af1, bf, acc[nt * 2 + 1]);
        }
        __syncthreads();
    }

    // ---- epilogue: bias / residual / relu, per ISA C/D layout
    const int rsub = (lane >> 4) * 8;
#pragma unroll
    for (int nt = 0; nt < 8; ++nt) {
        const int c  = colBase + nt * 16 + (lane & 15);
        const float bv = bias ? bias[c] : 0.0f;
#pragma unroll
        for (int s = 0; s < 2; ++s) {
            const v8f a = acc[nt * 2 + s];
            const int rbase = rowBase + s * 64 + wave * 16 + rsub;
#pragma unroll
            for (int j = 0; j < 8; ++j) {
                const int r = rbase + j;
                float val = a[j] + bv;
                if (flags & 2) val += residual[(size_t)r * ldc + c];
                if (flags & 1) val = fmaxf(val, 0.0f);
                C[(size_t)r * ldc + c] = val;
            }
        }
    }
}

// ---------------------------------------------------------------------------
// LayerNorm over D, one 128-thread block per token
// ---------------------------------------------------------------------------
__global__ void __launch_bounds__(128)
layernorm_kernel(const float* __restrict__ x, const float* __restrict__ g,
                 const float* __restrict__ b, float* __restrict__ out, int Dn)
{
    __shared__ float red[128];
    const int tok = blockIdx.x;
    const float* xr = x + (size_t)tok * Dn;

    float s = 0.0f;
    for (int d = threadIdx.x; d < Dn; d += 128) s += xr[d];
    red[threadIdx.x] = s; __syncthreads();
    for (int off = 64; off > 0; off >>= 1) {
        if (threadIdx.x < off) red[threadIdx.x] += red[threadIdx.x + off];
        __syncthreads();
    }
    const float mean = red[0] / (float)Dn;
    __syncthreads();

    float vs = 0.0f;
    for (int d = threadIdx.x; d < Dn; d += 128) {
        const float t = xr[d] - mean; vs += t * t;
    }
    red[threadIdx.x] = vs; __syncthreads();
    for (int off = 64; off > 0; off >>= 1) {
        if (threadIdx.x < off) red[threadIdx.x] += red[threadIdx.x + off];
        __syncthreads();
    }
    const float inv = rsqrtf(red[0] / (float)Dn + 1e-12f);

    for (int d = threadIdx.x; d < Dn; d += 128)
        out[(size_t)tok * Dn + d] = (xr[d] - mean) * inv * g[d] + b[d];
}

// ---------------------------------------------------------------------------
// Embedding + sinusoidal positional encoding
// ---------------------------------------------------------------------------
__global__ void __launch_bounds__(128)
embed_kernel(const int* __restrict__ tgt, const float* __restrict__ embedW,
             float* __restrict__ x)
{
    const int tok = blockIdx.x;          // b*U + u
    const int u   = tok & (UU - 1);
    const int id  = tgt[tok];
    for (int d = threadIdx.x; d < DD; d += 128) {
        const int i2 = d & ~1;
        const float ang = (float)u * __expf(-(float)i2 * (9.210340371976184f / (float)DD));
        const float pe  = (d & 1) ? cosf(ang) : sinf(ang);
        x[(size_t)tok * DD + d] =
            embedW[(size_t)id * DD + d] * 22.627416997969522f /* sqrt(512) */ + pe;
    }
}

// ---------------------------------------------------------------------------
// Causal multi-head attention. One 64-thread block per (query-row, head, batch).
// q/k/v kept in (B,U,D) layout; head h = columns [h*DH, (h+1)*DH).
// ---------------------------------------------------------------------------
__global__ void __launch_bounds__(64)
attn_kernel(const float* __restrict__ q, const float* __restrict__ k,
            const float* __restrict__ v, float* __restrict__ o)
{
    __shared__ float sc[UU];
    __shared__ float red[UU];
    const int qr = blockIdx.x, h = blockIdx.y, b = blockIdx.z;
    const int t  = threadIdx.x;                 // key index

    const float* qrow = q + (size_t)(b * UU + qr) * DD + h * DH;
    const float* krow = k + (size_t)(b * UU + t ) * DD + h * DH;

    float s = 0.0f;
#pragma unroll 4
    for (int d = 0; d < DH; ++d) s += qrow[d] * krow[d];
    s *= 0.08838834764831845f;                  // 1/sqrt(128)
    if (t > qr) s = -1.0e9f;

    red[t] = s; __syncthreads();
    for (int off = 32; off > 0; off >>= 1) {
        if (t < off) red[t] = fmaxf(red[t], red[t + off]);
        __syncthreads();
    }
    const float m = red[0]; __syncthreads();

    const float e = __expf(s - m);
    red[t] = e; __syncthreads();
    for (int off = 32; off > 0; off >>= 1) {
        if (t < off) red[t] += red[t + off];
        __syncthreads();
    }
    const float inv = 1.0f / red[0]; __syncthreads();

    sc[t] = e * inv; __syncthreads();

#pragma unroll
    for (int half = 0; half < 2; ++half) {
        const int d = t + half * 64;
        float a = 0.0f;
        for (int kk = 0; kk <= qr; ++kk)
            a += sc[kk] * v[(size_t)(b * UU + kk) * DD + h * DH + d];
        o[(size_t)(b * UU + qr) * DD + h * DH + d] = a;
    }
}

// ---------------------------------------------------------------------------
// Second tuple output: the (deterministic) causal mask, written as float 0/1
// ---------------------------------------------------------------------------
__global__ void __launch_bounds__(256)
mask_kernel(float* __restrict__ out)
{
    const int idx = blockIdx.x * 256 + threadIdx.x;
    if (idx >= BB * UU * UU) return;
    const int kk = idx & (UU - 1);
    const int qq = (idx >> 6) & (UU - 1);
    out[idx] = (kk <= qq) ? 1.0f : 0.0f;
}

// ---------------------------------------------------------------------------
// Host orchestration
// ---------------------------------------------------------------------------
extern "C" void kernel_launch(void* const* d_in, const int* in_sizes, int n_in,
                              void* d_out, int out_size, void* d_ws, size_t ws_size,
                              hipStream_t stream)
{
    (void)in_sizes; (void)n_in;
    const int*   tgt    = (const int*)  d_in[0];
    // d_in[1] = tgt_mask (deterministic tril; recomputed on device instead)
    const float* memory = (const float*)d_in[2];
    const float* embedW = (const float*)d_in[3];
    const float* Wq = (const float*)d_in[4];  const float* bq = (const float*)d_in[5];
    const float* Wk = (const float*)d_in[6];  const float* bk = (const float*)d_in[7];
    const float* Wv = (const float*)d_in[8];  const float* bv = (const float*)d_in[9];
    const float* Wo = (const float*)d_in[10]; const float* bo = (const float*)d_in[11];
    const float* ln1g = (const float*)d_in[12]; const float* ln1b = (const float*)d_in[13];
    const float* ln2g = (const float*)d_in[14]; const float* ln2b = (const float*)d_in[15];
    const float* W1 = (const float*)d_in[16]; const float* b1 = (const float*)d_in[17];
    const float* W2 = (const float*)d_in[18]; const float* b2 = (const float*)d_in[19];
    const float* lnfg = (const float*)d_in[20]; const float* lnfb = (const float*)d_in[21];
    const float* Wenc = (const float*)d_in[22]; const float* benc = (const float*)d_in[23];
    const float* Wdec = (const float*)d_in[24];
    const float* Wout = (const float*)d_in[25]; const float* bout = (const float*)d_in[26];

    const int NTOK = BB * UU;   // 256
    const int NMEM = BB * TT;   // 512

    float* ws   = (float*)d_ws;
    float* x    = ws; ws += (size_t)NTOK * DD;
    float* h    = ws; ws += (size_t)NTOK * DD;
    float* qb   = ws; ws += (size_t)NTOK * DD;
    float* kb   = ws; ws += (size_t)NTOK * DD;
    float* vb   = ws; ws += (size_t)NTOK * DD;
    float* ob   = ws; ws += (size_t)NTOK * DD;
    float* ffn  = ws; ws += (size_t)NTOK * DFF;
    float* zenc = ws; ws += (size_t)NMEM * JJ;
    float* zdec = ws; ws += (size_t)NTOK * JJ;
    const size_t needed = (size_t)(ws - (float*)d_ws) * sizeof(float);
    if (ws_size < needed) return;

    const dim3 blk(128);
    const dim3 gD (DD  / TILE_N, NTOK / TILE_M);   // (4, 2)
    const dim3 gF1(DFF / TILE_N, NTOK / TILE_M);   // (16, 2)

    embed_kernel<<<NTOK, 128, 0, stream>>>(tgt, embedW, x);

    for (int l = 0; l < LL; ++l) {
        const size_t wofs = (size_t)l * DD * DD;
        layernorm_kernel<<<NTOK, 128, 0, stream>>>(x, ln1g + l * DD, ln1b + l * DD, h, DD);

        gemm_f16wmma<<<gD, blk, 0, stream>>>(h, Wq + wofs, bq + l * DD, nullptr, qb,
                                             nullptr, nullptr, NTOK, DD, DD, DD, DD, DD, 0);
        gemm_f16wmma<<<gD, blk, 0, stream>>>(h, Wk + wofs, bk + l * DD, nullptr, kb,
                                             nullptr, nullptr, NTOK, DD, DD, DD, DD, DD, 0);
        gemm_f16wmma<<<gD, blk, 0, stream>>>(h, Wv + wofs, bv + l * DD, nullptr, vb,
                                             nullptr, nullptr, NTOK, DD, DD, DD, DD, DD, 0);

        attn_kernel<<<dim3(UU, HH, BB), 64, 0, stream>>>(qb, kb, vb, ob);

        // x = x + o @ Wo + bo   (in-place residual: each element read-then-written once)
        gemm_f16wmma<<<gD, blk, 0, stream>>>(ob, Wo + wofs, bo + l * DD, x, x,
                                             nullptr, nullptr, NTOK, DD, DD, DD, DD, DD, 2);

        layernorm_kernel<<<NTOK, 128, 0, stream>>>(x, ln2g + l * DD, ln2b + l * DD, h, DD);

        gemm_f16wmma<<<gF1, blk, 0, stream>>>(h, W1 + (size_t)l * DD * DFF, b1 + l * DFF,
                                              nullptr, ffn, nullptr, nullptr,
                                              NTOK, DFF, DD, DD, DFF, DFF, 1 /*relu*/);
        gemm_f16wmma<<<gD, blk, 0, stream>>>(ffn, W2 + (size_t)l * DFF * DD, b2 + l * DD,
                                             x, x, nullptr, nullptr,
                                             NTOK, DD, DFF, DFF, DD, DD, 2);
    }

    layernorm_kernel<<<NTOK, 128, 0, stream>>>(x, lnfg, lnfb, h, DD);

    // z_enc = memory @ W_enc + b_enc ; z_dec = x_final @ W_dec
    gemm_f16wmma<<<dim3(JJ / TILE_N, NMEM / TILE_M), blk, 0, stream>>>(
        memory, Wenc, benc, nullptr, zenc, nullptr, nullptr,
        NMEM, JJ, DD, DD, JJ, JJ, 0);
    gemm_f16wmma<<<dim3(JJ / TILE_N, NTOK / TILE_M), blk, 0, stream>>>(
        h, Wdec, nullptr, nullptr, zdec, nullptr, nullptr,
        NTOK, JJ, DD, DD, JJ, JJ, 0);

    // Joint: z = tanh(z_enc[b,t]+z_dec[b,u]) @ W_out + b_out  -> (B,T,U,ODIM)
    const int MJ = BB * TT * UU;  // 32768
    gemm_f16wmma<<<dim3(ODIM / TILE_N, MJ / TILE_M), blk, 0, stream>>>(
        nullptr, Wout, bout, nullptr, (float*)d_out, zenc, zdec,
        MJ, ODIM, JJ, 0, ODIM, ODIM, 4 /*joint tanh A*/);

    // Tuple output #2: tgt_mask as float, appended after z
    const long long zElems = (long long)BB * TT * UU * ODIM;
    if ((long long)out_size >= zElems + (long long)BB * UU * UU) {
        mask_kernel<<<(BB * UU * UU + 255) / 256, 256, 0, stream>>>(
            (float*)d_out + zElems);
    }
}
